// LayerStacks_83485574299885
// MI455X (gfx1250) — matrix operand order repaired and checked by reading, not compiled, not run on hardware
//
#include <hip/hip_runtime.h>
#include <stdint.h>

#define NCOUNT 8
#define NL1    3072
#define NKQ    4                 // K split factor
#define NKLEN  (NL1 / NKQ)       // 768 per wave
#define NL2    15
#define NL3    32
#define NB     16384
#define SQR_CORR (127.0f / 128.0f)

typedef __attribute__((ext_vector_type(16))) __bf16 v16bf;
typedef __attribute__((ext_vector_type(2)))  __bf16 v2bf;
typedef __attribute__((ext_vector_type(8)))  float  v8f;

struct U4 { uint32_t x, y, z, w; };
struct F4 { float a, b, c, d; };

union ABv {
    v16bf    h;
    uint32_t u[8];
    U4       q[2];
};

// fp32 pair -> packed bf16 dword (lo in [15:0], hi in [31:16]).
__device__ __forceinline__ uint32_t pk_bf16(float f0, float f1) {
#if __has_builtin(__builtin_amdgcn_cvt_pk_bf16_f32)
    union { v2bf v; uint32_t u; } cv;
    cv.v = __builtin_amdgcn_cvt_pk_bf16_f32(f0, f1);
    return cv.u;
#else
    union { v2bf v; uint32_t u; } cv;
    cv.v[0] = (__bf16)f0;
    cv.v[1] = (__bf16)f1;
    return cv.u;
#endif
}

__device__ __forceinline__ float clamp01(float v) {
    return fminf(fmaxf(v, 0.0f), 1.0f);
}

// ---------------- Prologue: fuse + convert W1 to bf16 --------------------
// Wb[k][n][K] = bf16( W1_stack[k][n][K] + W1_factor[n][K] ); 786 KB, L2-resident.
__global__ void fuse_w1_kernel(const float* __restrict__ Ws,
                               const float* __restrict__ Wf,
                               uint16_t* __restrict__ Wb) {
    int i = blockIdx.x * blockDim.x + threadIdx.x;
    if (i >= NCOUNT * 16 * NL1) return;
    int nk = i % (16 * NL1);
    uint32_t p = pk_bf16(Ws[i] + Wf[nk], 0.0f);
    Wb[i] = (uint16_t)(p & 0xFFFFu);
}

struct XRegs { F4 a, b, c, d; };   // 16 fp32 K-values for one tile, this lane

__device__ __forceinline__ void load_x(const float* base, XRegs& r) {
    const F4* p  = reinterpret_cast<const F4*>(base);
    const F4* p2 = reinterpret_cast<const F4*>(base + 16);
    r.a = p[0]; r.b = p[1]; r.c = p2[0]; r.d = p2[1];
}

__device__ __forceinline__ void cvt_tile(const XRegs& r, ABv& a) {
    a.u[0] = pk_bf16(r.a.a, r.a.b); a.u[1] = pk_bf16(r.a.c, r.a.d);
    a.u[2] = pk_bf16(r.b.a, r.b.b); a.u[3] = pk_bf16(r.b.c, r.b.d);
    a.u[4] = pk_bf16(r.c.a, r.c.b); a.u[5] = pk_bf16(r.c.c, r.c.d);
    a.u[6] = pk_bf16(r.d.a, r.d.b); a.u[7] = pk_bf16(r.d.c, r.d.d);
}

// ---------------- Layer-1 GEMM, K-split 4 ways ---------------------------
// grid = (NB/128, NKQ); each wave owns 32 samples x one 768-wide K-quarter.
// Per 32-wide K step: cvt x fp32->bf16 A operand; for each stack k: load
// fused bf16 B, AND/cndmask-zero the A rows whose ls_index != k, accumulate
// with v_wmma_f32_16x16x32_bf16 (masked sum over k == per-row weight gather).
// fp32 partial accumulators go to d_ws; a tail kernel reduces the 4 quarters.
__global__ __launch_bounds__(128) void l1_gemm_kernel(
    const float* __restrict__ x,
    const int*   __restrict__ lsi,
    const uint16_t* __restrict__ Wb,
    float* __restrict__ part) {

    const int lane  = threadIdx.x & 31;
    const int wave  = threadIdx.x >> 5;
    const int sbase = blockIdx.x * 128 + wave * 32;
    const int kq    = blockIdx.y;
    const int kbeg  = kq * NKLEN;

    const int rowA  = lane & 15;       // A row / B column owned by this lane
    const int hi    = lane >> 4;       // lane group (K-half select)
    const int koffA = hi ? 8  : 0;     // A: lanes 16-31 hold K=8..15,24..31
    const int koffB = hi ? 8  : 0;     // B: lanes 16-31 hold K=8..15,24..31

    const int idx0 = lsi[sbase + rowA];
    const int idx1 = lsi[sbase + 16 + rowA];

    // Per-stack lane-select masks (compiler hoists to SGPR lane-masks and
    // emits v_dual_cndmask in the hot loop).
    uint32_t sel0[NCOUNT], sel1[NCOUNT];
    #pragma unroll
    for (int k = 0; k < NCOUNT; ++k) {
        sel0[k] = (idx0 == k) ? 0xFFFFFFFFu : 0u;
        sel1[k] = (idx1 == k) ? 0xFFFFFFFFu : 0u;
    }

    v8f acc0 = {};
    v8f acc1 = {};

    const float* x0 = x + (size_t)(sbase + rowA) * NL1 + kbeg + koffA;
    const float* x1 = x + (size_t)(sbase + 16 + rowA) * NL1 + kbeg + koffA;

    // Software pipeline: next K-step's fp32 x stays in flight during WMMAs;
    // prefetch runs 8 K-steps (~1 us) ahead to cover HBM latency.
    XRegs c0, c1, n0, n1;
    load_x(x0, c0);
    load_x(x1, c1);

    // Ping-pong masked-operand buffers (by k parity) so the WAR overwrite of
    // a WMMA's A operand is >= 2 WMMAs + 8 cndmasks away: no v_nop hazards.
    ABv m0buf[2], m1buf[2];

    for (int kk = 0; kk < NKLEN; kk += 32) {
        const int kn = kk + 32;
        if (kn < NKLEN) {
            load_x(x0 + kn, n0);
            load_x(x1 + kn, n1);
        }
        __builtin_prefetch(x0 + kk + 256, 0, 1);
        __builtin_prefetch(x1 + kk + 256, 0, 1);

        ABv a0, a1;
        cvt_tile(c0, a0);
        cvt_tile(c1, a1);

        #pragma unroll
        for (int k = 0; k < NCOUNT; ++k) {
            const uint16_t* wp =
                Wb + (size_t)(k * 16 + rowA) * NL1 + kbeg + kk + koffB;
            ABv bv;
            bv.q[0] = *reinterpret_cast<const U4*>(wp);
            bv.q[1] = *reinterpret_cast<const U4*>(wp + 16);

            ABv& m0 = m0buf[k & 1];
            #pragma unroll
            for (int i = 0; i < 8; ++i) m0.u[i] = a0.u[i] & sel0[k];
            acc0 = __builtin_amdgcn_wmma_f32_16x16x32_bf16(
                false, m0.h, false, bv.h, (short)0, acc0, false, false);

            ABv& m1 = m1buf[k & 1];
            #pragma unroll
            for (int i = 0; i < 8; ++i) m1.u[i] = a1.u[i] & sel1[k];
            acc1 = __builtin_amdgcn_wmma_f32_16x16x32_bf16(
                false, m1.h, false, bv.h, (short)0, acc1, false, false);
        }

        c0 = n0;
        c1 = n1;
    }

    // Spill fp32 partials: part[kq][sample][n].
    // D layout: VGPR r, lanes 0-15 -> M=r, lanes 16-31 -> M=8+r; N = lane&15.
    float* pp = part + ((size_t)kq * NB + sbase) * 16;
    #pragma unroll
    for (int r = 0; r < 8; ++r) {
        const int m0 = r + hi * 8;
        pp[m0 * 16 + rowA]        = acc0[r];
        pp[(16 + m0) * 16 + rowA] = acc1[r];
    }
}

// ---------------- Tail: reduce K-quarters + small layers -----------------
__global__ __launch_bounds__(256) void tail_kernel(
    const int*   __restrict__ lsi,
    const float* __restrict__ part,
    const float* __restrict__ b1,
    const float* __restrict__ W2,
    const float* __restrict__ b2,
    const float* __restrict__ W3,
    const float* __restrict__ b3,
    float* __restrict__ out) {

    const int sample = blockIdx.x * blockDim.x + threadIdx.x;
    if (sample >= NB) return;
    const int id = lsi[sample];

    float l1c[16];
    #pragma unroll
    for (int n = 0; n < 16; ++n) l1c[n] = b1[id * 16 + n];
    #pragma unroll
    for (int q = 0; q < NKQ; ++q) {
        const float* pq = part + ((size_t)q * NB + sample) * 16;
        #pragma unroll
        for (int n = 0; n < 16; ++n) l1c[n] += pq[n];
    }

    const float l1x_out = l1c[NL2];

    float h[2 * NL2];
    #pragma unroll
    for (int j = 0; j < NL2; ++j) {
        const float v = l1c[j];
        h[j]       = clamp01(v * v * SQR_CORR);
        h[NL2 + j] = clamp01(v);
    }

    const float* w2 = W2 + (size_t)id * NL3 * (2 * NL2);
    float s3 = b3[id];
    const float* w3 = W3 + (size_t)id * NL3;
    for (int n2 = 0; n2 < NL3; ++n2) {
        float s = b2[id * NL3 + n2];
        const float* wr = w2 + n2 * (2 * NL2);
        #pragma unroll
        for (int j = 0; j < 2 * NL2; ++j) s += wr[j] * h[j];
        s3 += w3[n2] * clamp01(s);
    }

    out[sample] = s3 + l1x_out;
}

extern "C" void kernel_launch(void* const* d_in, const int* in_sizes, int n_in,
                              void* d_out, int out_size, void* d_ws, size_t ws_size,
                              hipStream_t stream) {
    (void)in_sizes; (void)n_in; (void)out_size; (void)ws_size;

    const float* x   = (const float*)d_in[0];
    const int*   lsi = (const int*)d_in[1];
    const float* W1s = (const float*)d_in[2];
    const float* W1f = (const float*)d_in[3];
    const float* b1  = (const float*)d_in[4];
    const float* W2  = (const float*)d_in[5];
    const float* b2  = (const float*)d_in[6];
    const float* W3  = (const float*)d_in[7];
    const float* b3  = (const float*)d_in[8];
    float* out = (float*)d_out;

    // Workspace layout: [0, 786432) fused bf16 W1; then 4 MB of fp32 partials.
    uint16_t* Wb   = (uint16_t*)d_ws;
    float*    part = (float*)((char*)d_ws + (size_t)NCOUNT * 16 * NL1 * 2);

    const int fuse_elems = NCOUNT * 16 * NL1;
    fuse_w1_kernel<<<(fuse_elems + 255) / 256, 256, 0, stream>>>(W1s, W1f, Wb);

    dim3 ggrid(NB / 128, NKQ);     // 128 x 4 blocks, 4 waves each -> 2048 waves
    l1_gemm_kernel<<<ggrid, 128, 0, stream>>>(x, lsi, Wb, part);

    tail_kernel<<<(NB + 255) / 256, 256, 0, stream>>>(
        lsi, part, b1, W2, b2, W3, b3, out);
}
